// LatentODEModel_19121194401976
// MI455X (gfx1250) — matrix operand order repaired and checked by reading, not compiled, not run on hardware
//
#include <hip/hip_runtime.h>
#include <hip/hip_bf16.h>
#include <math.h>

// ---------------------------------------------------------------------------
// Latent ODE (encoder -> dopri5 -> decoder) for MI455X / gfx1250, wave32.
//   B=512, T=512, D_IN=128, D_LAT=8, D_HID=32
// Output: preds [B*T*128] f32  ++  traj [B*T*8] f32 (concatenated in d_out).
// ---------------------------------------------------------------------------

#define NB    512
#define NT    512
#define DIN   128
#define DLAT  8
#define DHID  32

typedef __attribute__((ext_vector_type(2)))  float     v2f;
typedef __attribute__((ext_vector_type(8)))  float     v8f;
typedef __attribute__((ext_vector_type(8)))  _Float16  v8h;
typedef __attribute__((ext_vector_type(16))) _Float16  v16h;

// Decoder weight fragments, pre-swizzled into WMMA 16-bit B layout:
//   fragment f, lane l (n = l&15, half = l>>4), element e -> K = 8*half + e (e<8), 16 + 8*half + (e-8) (e>=8).
//   frag 0,1   : dec W1 (8x32, zero-padded to K=32), n0 = 0,16
//   frag 2,3   : dec W2 (32x32),                     n0 = 0,16
//   frag 4..11 : dec W3 (32x128),                    n0 = 16*(f-4)
#define NFRAG 12
#define FRAG_HALVES 512   /* 32 lanes x 16 halves */

// ===========================================================================
// Kernel 0: swizzle decoder weights to f16 fragment layout in workspace.
// ===========================================================================
__global__ void prep_weights_kernel(const float* __restrict__ dW1,
                                    const float* __restrict__ dW2,
                                    const float* __restrict__ dW3,
                                    _Float16* __restrict__ wsw) {
  int tid = threadIdx.x + blockIdx.x * blockDim.x;
  int stride = blockDim.x * gridDim.x;
  for (int idx = tid; idx < NFRAG * FRAG_HALVES; idx += stride) {
    int f = idx >> 9;            // fragment
    int l = (idx >> 4) & 31;     // lane
    int e = idx & 15;            // element
    int half = l >> 4, n = l & 15;
    int K = (e < 8) ? (8 * half + e) : (16 + 8 * half + (e - 8));
    float v;
    if (f < 2)       v = (K < DLAT) ? dW1[K * DHID + f * 16 + n] : 0.0f;
    else if (f < 4)  v = dW2[K * DHID + (f - 2) * 16 + n];
    else             v = dW3[K * DIN + (f - 4) * 16 + n];
    wsw[idx] = (_Float16)v;
  }
}

// ===========================================================================
// Kernel 1: encoder, f32 WMMA (V_WMMA_F32_16X16X4_F32), 16 rows per wave.
// ===========================================================================
__global__ __launch_bounds__(32) void encode_kernel(
    const float* __restrict__ X,
    const float* __restrict__ W1, const float* __restrict__ b1,
    const float* __restrict__ W2, const float* __restrict__ b2,
    const float* __restrict__ W3, const float* __restrict__ b3,
    float* __restrict__ z0) {
  __shared__ float hbuf[16 * 33];
  const int lane = threadIdx.x;
  const int r0 = blockIdx.x * 16;
  const int m = lane & 15;
  const int half = lane >> 4;
  const int kh = half * 2;
  const int n = lane & 15;

  // ---- layer 1: X[16x128] @ W1[128x32] + b1, ReLU -> LDS ----
  v8f c0 = {}, c1 = {};
  for (int k0 = 0; k0 < DIN; k0 += 4) {
    v2f a, bA, bB;
    a.x  = X[(size_t)(r0 + m) * DIN + k0 + kh];
    a.y  = X[(size_t)(r0 + m) * DIN + k0 + kh + 1];
    bA.x = W1[(k0 + kh) * DHID + n];
    bA.y = W1[(k0 + kh + 1) * DHID + n];
    bB.x = W1[(k0 + kh) * DHID + 16 + n];
    bB.y = W1[(k0 + kh + 1) * DHID + 16 + n];
    c0 = __builtin_amdgcn_wmma_f32_16x16x4_f32(false, a, false, bA, (short)0, c0, false, false);
    c1 = __builtin_amdgcn_wmma_f32_16x16x4_f32(false, a, false, bB, (short)0, c1, false, false);
  }
#pragma unroll
  for (int r = 0; r < 8; r++) {
    int mr = r + 8 * half;
    hbuf[mr * 33 + n]      = fmaxf(c0[r] + b1[n], 0.0f);
    hbuf[mr * 33 + 16 + n] = fmaxf(c1[r] + b1[16 + n], 0.0f);
  }
  __syncthreads();

  // ---- layer 2: h1[16x32] @ W2[32x32] + b2, ReLU -> LDS ----
  v8f d0 = {}, d1 = {};
  for (int k0 = 0; k0 < DHID; k0 += 4) {
    v2f a, bA, bB;
    a.x  = hbuf[m * 33 + k0 + kh];
    a.y  = hbuf[m * 33 + k0 + kh + 1];
    bA.x = W2[(k0 + kh) * DHID + n];
    bA.y = W2[(k0 + kh + 1) * DHID + n];
    bB.x = W2[(k0 + kh) * DHID + 16 + n];
    bB.y = W2[(k0 + kh + 1) * DHID + 16 + n];
    d0 = __builtin_amdgcn_wmma_f32_16x16x4_f32(false, a, false, bA, (short)0, d0, false, false);
    d1 = __builtin_amdgcn_wmma_f32_16x16x4_f32(false, a, false, bB, (short)0, d1, false, false);
  }
  __syncthreads();
#pragma unroll
  for (int r = 0; r < 8; r++) {
    int mr = r + 8 * half;
    hbuf[mr * 33 + n]      = fmaxf(d0[r] + b2[n], 0.0f);
    hbuf[mr * 33 + 16 + n] = fmaxf(d1[r] + b2[16 + n], 0.0f);
  }
  __syncthreads();

  // ---- layer 3 (linear): h2[16x32] @ W3[32x8] + b3 -> z0 ----
  v8f e0 = {};
  for (int k0 = 0; k0 < DHID; k0 += 4) {
    v2f a, bA;
    a.x  = hbuf[m * 33 + k0 + kh];
    a.y  = hbuf[m * 33 + k0 + kh + 1];
    bA.x = (n < DLAT) ? W3[(k0 + kh) * DLAT + n] : 0.0f;
    bA.y = (n < DLAT) ? W3[(k0 + kh + 1) * DLAT + n] : 0.0f;
    e0 = __builtin_amdgcn_wmma_f32_16x16x4_f32(false, a, false, bA, (short)0, e0, false, false);
  }
  if (n < DLAT) {
#pragma unroll
    for (int r = 0; r < 8; r++) {
      int mr = r + 8 * half;
      z0[(size_t)(r0 + mr) * DLAT + n] = e0[r] + b3[n];
    }
  }
}

// ===========================================================================
// Kernel 2: adaptive dopri5, one wave per sample. lanes = hidden units.
// ===========================================================================
__device__ __forceinline__ void dyn_eval(const float* z, float* f,
                                         const float* w1c, float b1l,
                                         const float* w2c, float b2l,
                                         const float* w3r, const float* b3) {
  float acc = b1l;
#pragma unroll
  for (int k = 0; k < DLAT; k++) acc = fmaf(z[k], w1c[k], acc);
  float h1 = tanhf(acc);

  float acc2 = b2l;
#pragma unroll
  for (int k = 0; k < DHID; k++) acc2 = fmaf(__shfl(h1, k, 32), w2c[k], acc2);
  float h2 = tanhf(acc2);

#pragma unroll
  for (int j = 0; j < DLAT; j++) {
    float p = h2 * w3r[j];
#pragma unroll
    for (int off = 16; off >= 1; off >>= 1) p += __shfl_xor(p, off, 32);
    f[j] = p + b3[j];
  }
}

__global__ __launch_bounds__(32) void integrate_kernel(
    const float* __restrict__ z0buf, const float* __restrict__ qt,
    const float* __restrict__ W1, const float* __restrict__ b1,
    const float* __restrict__ W2, const float* __restrict__ b2,
    const float* __restrict__ W3, const float* __restrict__ b3g,
    float* __restrict__ traj) {
  const int b = blockIdx.x;
  const int lane = threadIdx.x;

  float w1c[DLAT], w2c[DHID], w3r[DLAT], b3[DLAT];
#pragma unroll
  for (int k = 0; k < DLAT; k++) w1c[k] = W1[k * DHID + lane];
  float b1l = b1[lane];
#pragma unroll
  for (int k = 0; k < DHID; k++) w2c[k] = W2[k * DHID + lane];
  float b2l = b2[lane];
#pragma unroll
  for (int j = 0; j < DLAT; j++) { w3r[j] = W3[lane * DLAT + j]; b3[j] = b3g[j]; }

  float z[DLAT];
#pragma unroll
  for (int j = 0; j < DLAT; j++) z[j] = z0buf[(size_t)b * DLAT + j];

  const size_t qbase = (size_t)b * NT;
  float t = qt[qbase];

  if (lane == 0) {
    float4 lo = {z[0], z[1], z[2], z[3]}, hi = {z[4], z[5], z[6], z[7]};
    *(float4*)&traj[qbase * DLAT]     = lo;
    *(float4*)&traj[qbase * DLAT + 4] = hi;
  }

  float k1[DLAT];
  dyn_eval(z, k1, w1c, b1l, w2c, b2l, w3r, b3);
  float h = 1e-2f;

  const float A21 = 0.2f;
  const float A31 = 3.0f/40.0f, A32 = 9.0f/40.0f;
  const float A41 = 44.0f/45.0f, A42 = -56.0f/15.0f, A43 = 32.0f/9.0f;
  const float A51 = 19372.0f/6561.0f, A52 = -25360.0f/2187.0f,
              A53 = 64448.0f/6561.0f, A54 = -212.0f/729.0f;
  const float A61 = 9017.0f/3168.0f, A62 = -355.0f/33.0f,
              A63 = 46732.0f/5247.0f, A64 = 49.0f/176.0f, A65 = -5103.0f/18656.0f;
  const float B1 = 35.0f/384.0f, B3 = 500.0f/1113.0f, B4 = 125.0f/192.0f,
              B5 = -2187.0f/6784.0f, B6 = 11.0f/84.0f;
  const float E1 = 71.0f/57600.0f, E3 = -71.0f/16695.0f, E4 = 71.0f/1920.0f,
              E5 = -17253.0f/339200.0f, E6 = 22.0f/525.0f, E7 = -1.0f/40.0f;
  const float RTOL = 1e-7f, ATOL = 1e-9f;

  for (int i = 1; i < NT; i++) {
    float tn = qt[qbase + i];
    int guard = 0;
    while (t < tn - 1e-9f && guard < 100000) {
      guard++;
      float hs = fminf(h, tn - t);
      float yt[DLAT], k2[DLAT], k3[DLAT], k4[DLAT], k5[DLAT], k6[DLAT], k7[DLAT], y5[DLAT];

#pragma unroll
      for (int j = 0; j < DLAT; j++) yt[j] = z[j] + hs * (A21 * k1[j]);
      dyn_eval(yt, k2, w1c, b1l, w2c, b2l, w3r, b3);
#pragma unroll
      for (int j = 0; j < DLAT; j++) yt[j] = z[j] + hs * (A31 * k1[j] + A32 * k2[j]);
      dyn_eval(yt, k3, w1c, b1l, w2c, b2l, w3r, b3);
#pragma unroll
      for (int j = 0; j < DLAT; j++) yt[j] = z[j] + hs * (A41 * k1[j] + A42 * k2[j] + A43 * k3[j]);
      dyn_eval(yt, k4, w1c, b1l, w2c, b2l, w3r, b3);
#pragma unroll
      for (int j = 0; j < DLAT; j++)
        yt[j] = z[j] + hs * (A51 * k1[j] + A52 * k2[j] + A53 * k3[j] + A54 * k4[j]);
      dyn_eval(yt, k5, w1c, b1l, w2c, b2l, w3r, b3);
#pragma unroll
      for (int j = 0; j < DLAT; j++)
        yt[j] = z[j] + hs * (A61 * k1[j] + A62 * k2[j] + A63 * k3[j] + A64 * k4[j] + A65 * k5[j]);
      dyn_eval(yt, k6, w1c, b1l, w2c, b2l, w3r, b3);
#pragma unroll
      for (int j = 0; j < DLAT; j++)
        y5[j] = z[j] + hs * (B1 * k1[j] + B3 * k3[j] + B4 * k4[j] + B5 * k5[j] + B6 * k6[j]);
      dyn_eval(y5, k7, w1c, b1l, w2c, b2l, w3r, b3);   // FSAL

      float s = 0.0f;
#pragma unroll
      for (int j = 0; j < DLAT; j++) {
        float e = hs * (E1 * k1[j] + E3 * k3[j] + E4 * k4[j] + E5 * k5[j] + E6 * k6[j] + E7 * k7[j]);
        float sc = ATOL + RTOL * fmaxf(fabsf(z[j]), fabsf(y5[j]));
        float q = e / sc;
        s += q * q;
      }
      float errn = sqrtf(s * (1.0f / DLAT));

      if (errn <= 1.0f || hs <= 1e-7f) {
        t += hs;
#pragma unroll
        for (int j = 0; j < DLAT; j++) { z[j] = y5[j]; k1[j] = k7[j]; }
      }
      float fac = 0.9f * powf(fmaxf(errn, 1e-12f), -0.2f);
      fac = fminf(fmaxf(fac, 0.2f), 5.0f);
      h = hs * fac;
    }
    if (lane == 0) {
      float4 lo = {z[0], z[1], z[2], z[3]}, hi = {z[4], z[5], z[6], z[7]};
      *(float4*)&traj[(qbase + i) * DLAT]     = lo;
      *(float4*)&traj[(qbase + i) * DLAT + 4] = hi;
    }
  }
}

// ===========================================================================
// Kernel 3: decoder. f16-in/f32-acc WMMA 16x16x32. 16 rows per wave, 8 waves
// per block. Weights async-staged into LDS once per block; B fragments are
// one contiguous 32B v16h load per lane; A fragments are two 16B LDS loads.
//   16-bit A 16x32 layout: lane(l): M=l&15, half=l>>4,
//     elements 0..7  -> K =  8*half + 0..7   (contiguous)
//     elements 8..15 -> K = 16 + 8*half + 0..7 (contiguous)
// ===========================================================================
#define HSTR 40   /* LDS row stride in halves: 80B, keeps 16B alignment */

__global__ __launch_bounds__(256) void decode_kernel(
    const float* __restrict__ traj,
    const _Float16* __restrict__ wsw,
    const float* __restrict__ db1, const float* __restrict__ db2,
    const float* __restrict__ db3,
    const unsigned char* __restrict__ mask,
    float* __restrict__ preds) {
  __shared__ __align__(16) _Float16 wsm[NFRAG * FRAG_HALVES];   // 12 KB
  __shared__ __align__(16) _Float16 hall[8][16 * HSTR];         // 10 KB

  // ---- async-stage swizzled weights (12 KB) into LDS, once per block ----
  {
    unsigned long long gbase = (unsigned long long)(size_t)wsw;
    unsigned int lbase = (unsigned int)(size_t)(void*)wsm;   // LDS offset (addr[31:0])
#pragma unroll
    for (int i = 0; i < 3; i++) {
      unsigned int c = threadIdx.x + i * 256u;                // 16B chunk id
      unsigned long long g = gbase + (unsigned long long)c * 16ull;
      unsigned int l = lbase + c * 16u;
      asm volatile("global_load_async_to_lds_b128 %0, %1, off"
                   :: "v"(l), "v"(g) : "memory");
    }
    asm volatile("s_wait_asynccnt 0x0" ::: "memory");
  }
  __syncthreads();

  const int wv = threadIdx.x >> 5;
  const int lane = threadIdx.x & 31;
  const int r0 = (blockIdx.x * 8 + wv) * 16;   // row into flattened [B*T]
  _Float16* h = hall[wv];
  const int m = lane & 15, half = lane >> 4, n = lane & 15;
  const _Float16* wl = &wsm[lane * 16];        // + f*FRAG_HALVES per fragment

  // ---- layer 1: traj[16x8] (K-padded to 32) @ W1 ----
  v16h a1 = {};
  if (half == 0) {   // only K=0..7 are nonzero, all owned by lanes 0..15
    const float4 t0 = *(const float4*)&traj[(size_t)(r0 + m) * DLAT];
    const float4 t1 = *(const float4*)&traj[(size_t)(r0 + m) * DLAT + 4];
    a1[0] = (_Float16)t0.x; a1[1] = (_Float16)t0.y;
    a1[2] = (_Float16)t0.z; a1[3] = (_Float16)t0.w;
    a1[4] = (_Float16)t1.x; a1[5] = (_Float16)t1.y;
    a1[6] = (_Float16)t1.z; a1[7] = (_Float16)t1.w;
  }
  v16h bA = *(const v16h*)&wl[0 * FRAG_HALVES];
  v16h bB = *(const v16h*)&wl[1 * FRAG_HALVES];
  v8f c0 = {}, c1 = {};
  c0 = __builtin_amdgcn_wmma_f32_16x16x32_f16(false, a1, false, bA, (short)0, c0, false, false);
  c1 = __builtin_amdgcn_wmma_f32_16x16x32_f16(false, a1, false, bB, (short)0, c1, false, false);
#pragma unroll
  for (int r = 0; r < 8; r++) {
    int mr = r + 8 * half;
    h[mr * HSTR + n]      = (_Float16)fmaxf(c0[r] + db1[n], 0.0f);
    h[mr * HSTR + 16 + n] = (_Float16)fmaxf(c1[r] + db1[16 + n], 0.0f);
  }
  __syncthreads();

  // ---- layer 2: h1[16x32] @ W2 ----
  v8h lo2 = *(const v8h*)&h[m * HSTR + 8 * half];
  v8h hi2 = *(const v8h*)&h[m * HSTR + 16 + 8 * half];
  v16h a2 = __builtin_shufflevector(lo2, hi2, 0,1,2,3,4,5,6,7,8,9,10,11,12,13,14,15);
  v16h b2A = *(const v16h*)&wl[2 * FRAG_HALVES];
  v16h b2B = *(const v16h*)&wl[3 * FRAG_HALVES];
  v8f d0 = {}, d1 = {};
  d0 = __builtin_amdgcn_wmma_f32_16x16x32_f16(false, a2, false, b2A, (short)0, d0, false, false);
  d1 = __builtin_amdgcn_wmma_f32_16x16x32_f16(false, a2, false, b2B, (short)0, d1, false, false);
  __syncthreads();
#pragma unroll
  for (int r = 0; r < 8; r++) {
    int mr = r + 8 * half;
    h[mr * HSTR + n]      = (_Float16)fmaxf(d0[r] + db2[n], 0.0f);
    h[mr * HSTR + 16 + n] = (_Float16)fmaxf(d1[r] + db2[16 + n], 0.0f);
  }
  __syncthreads();

  // ---- layer 3: h2[16x32] @ W3[32x128], ReLU, mask, store ----
  v8h lo3 = *(const v8h*)&h[m * HSTR + 8 * half];
  v8h hi3 = *(const v8h*)&h[m * HSTR + 16 + 8 * half];
  v16h a3 = __builtin_shufflevector(lo3, hi3, 0,1,2,3,4,5,6,7,8,9,10,11,12,13,14,15);
#pragma unroll
  for (int t3 = 0; t3 < 8; t3++) {
    v16h bW = *(const v16h*)&wl[(4 + t3) * FRAG_HALVES];
    v8f c = {};
    c = __builtin_amdgcn_wmma_f32_16x16x32_f16(false, a3, false, bW, (short)0, c, false, false);
#pragma unroll
    for (int r = 0; r < 8; r++) {
      int mr = r + 8 * half;
      int row = r0 + mr;
      int col = t3 * 16 + n;
      float v = fmaxf(c[r] + db3[col], 0.0f);
      v = mask[row] ? v : 0.0f;
      preds[(size_t)row * DIN + col] = v;
    }
  }
}

// ===========================================================================
// Host launcher
// ===========================================================================
extern "C" void kernel_launch(void* const* d_in, const int* in_sizes, int n_in,
                              void* d_out, int out_size, void* d_ws, size_t ws_size,
                              hipStream_t stream) {
  (void)in_sizes; (void)n_in; (void)out_size; (void)ws_size;

  const float* first_obs = (const float*)d_in[0];
  // d_in[1] = first_time: unused by the reference computation as well.
  const float* qt = (const float*)d_in[2];
  const unsigned char* mask = (const unsigned char*)d_in[3];  // jnp bool -> 1 byte
  const float *encW1 = (const float*)d_in[4],  *encb1 = (const float*)d_in[5];
  const float *encW2 = (const float*)d_in[6],  *encb2 = (const float*)d_in[7];
  const float *encW3 = (const float*)d_in[8],  *encb3 = (const float*)d_in[9];
  const float *dynW1 = (const float*)d_in[10], *dynb1 = (const float*)d_in[11];
  const float *dynW2 = (const float*)d_in[12], *dynb2 = (const float*)d_in[13];
  const float *dynW3 = (const float*)d_in[14], *dynb3 = (const float*)d_in[15];
  const float *decW1 = (const float*)d_in[16], *decb1 = (const float*)d_in[17];
  const float *decW2 = (const float*)d_in[18], *decb2 = (const float*)d_in[19];
  const float *decW3 = (const float*)d_in[20], *decb3 = (const float*)d_in[21];

  float* preds = (float*)d_out;
  float* traj  = preds + (size_t)NB * NT * DIN;   // second output region

  // workspace layout: z0 (f32), then swizzled f16 decoder weight fragments
  float* z0 = (float*)d_ws;                                        // 512*8 f32
  _Float16* wsw = (_Float16*)((char*)d_ws + NB * DLAT * sizeof(float));  // 12 KB

  prep_weights_kernel<<<8, 256, 0, stream>>>(decW1, decW2, decW3, wsw);
  encode_kernel<<<NB / 16, 32, 0, stream>>>(first_obs, encW1, encb1, encW2, encb2,
                                            encW3, encb3, z0);
  integrate_kernel<<<NB, 32, 0, stream>>>(z0, qt, dynW1, dynb1, dynW2, dynb2,
                                          dynW3, dynb3, traj);
  decode_kernel<<<(NB * NT / 16) / 8, 256, 0, stream>>>(traj, wsw, decb1, decb2,
                                                        decb3, mask, preds);
}
